// GatedShiftMixer_30434138259751
// MI455X (gfx1250) — compile-verified, matrix operand order checked by reading, not run on hardware
//
#include <hip/hip_runtime.h>

// GatedShiftMixer for MI455X (gfx1250, wave32).
// Strategy: bf16 WMMA (v_wmma_f32_16x16x32_bf16) for both GEMMs, gate GEMM fused
// with sigmoid*shifted-v mix and LayerNorm so the 576MB gates tensor is never
// materialized. gate_W (18MB bf16) and v (32MB bf16) stay L2-resident (192MB L2).

typedef __attribute__((ext_vector_type(16))) __bf16 v16bf;
typedef __attribute__((ext_vector_type(8)))  float  v8f;

#define DIM   1024
#define SEQ   4096
#define ROWS  16384          // B*L
#define NGRP  9

__device__ __forceinline__ unsigned short f2bf(float f) {
    unsigned int u = __float_as_uint(f);
    return (unsigned short)((u + 0x7FFFu + ((u >> 16) & 1u)) >> 16);  // RNE
}
__device__ __forceinline__ float bf2f(unsigned short h) {
    return __uint_as_float(((unsigned int)h) << 16);
}
__device__ __forceinline__ v8f zero8() {
    v8f z;
#pragma unroll
    for (int i = 0; i < 8; ++i) z[i] = 0.0f;
    return z;
}

// ---------------------------------------------------------------- convert ----
__global__ __launch_bounds__(256) void cvt_f32_bf16(const float* __restrict__ src,
                                                    unsigned short* __restrict__ dst,
                                                    int n4) {
    int i = blockIdx.x * blockDim.x + threadIdx.x;
    int stride = gridDim.x * blockDim.x;
    for (; i < n4; i += stride) {
        float4 f = reinterpret_cast<const float4*>(src)[i];
        ushort4 h;
        h.x = f2bf(f.x); h.y = f2bf(f.y); h.z = f2bf(f.z); h.w = f2bf(f.w);
        reinterpret_cast<ushort4*>(dst)[i] = h;
    }
}

// ------------------------------------------------------------ value GEMM ----
// v = x @ value_W^T ; one workgroup = 16 rows x 1024 cols, 8 waves x 128 cols.
__global__ __launch_bounds__(256) void value_gemm(const unsigned short* __restrict__ xbf,
                                                  const unsigned short* __restrict__ wv,
                                                  unsigned short* __restrict__ vbf) {
    __shared__ __align__(32) unsigned short lds_x[16 * DIM];   // 32 KB A band

    const int M0   = blockIdx.x * 16;
    const int lane = threadIdx.x & 31;
    const int wav  = threadIdx.x >> 5;     // 0..7
    const int l16  = lane & 15;
    const int kh   = lane >> 4;            // K-half selector
    const int N0   = wav * 128;

    // stage 16 contiguous rows of x (bf16) into LDS
    const uint4* gsrc = reinterpret_cast<const uint4*>(xbf + (size_t)M0 * DIM);
    for (int it = threadIdx.x; it < 2048; it += 256)
        reinterpret_cast<uint4*>(lds_x)[it] = gsrc[it];
    __syncthreads();

    v8f acc[8];
#pragma unroll
    for (int t = 0; t < 8; ++t) acc[t] = zero8();

    for (int k = 0; k < DIM; k += 32) {
        v16bf a = *reinterpret_cast<const v16bf*>(&lds_x[l16 * DIM + k + kh * 16]);
#pragma unroll
        for (int t = 0; t < 8; ++t) {
            int colB = N0 + t * 16 + l16;
            v16bf b = *reinterpret_cast<const v16bf*>(&wv[(size_t)colB * DIM + k + kh * 16]);
            acc[t] = __builtin_amdgcn_wmma_f32_16x16x32_bf16(
                         false, a, false, b, (short)0, acc[t], false, false);
        }
    }

#pragma unroll
    for (int t = 0; t < 8; ++t) {
        int col = N0 + t * 16 + l16;
#pragma unroll
        for (int i = 0; i < 8; ++i) {
            int row = M0 + i + kh * 8;
            vbf[(size_t)row * DIM + col] = f2bf(acc[t][i]);
        }
    }
}

// ---------------------------------------------- fused gate GEMM + mix + LN ---
// out[l,:] = LN( sum_j sigmoid(x[l] @ gate_W[j]^T) * v[(l+s_j) mod L] )
__global__ __launch_bounds__(256) void gate_mix_ln(const unsigned short* __restrict__ xbf,
                                                   const unsigned short* __restrict__ gw,
                                                   const unsigned short* __restrict__ vbf,
                                                   const float* __restrict__ gamma,
                                                   const float* __restrict__ beta,
                                                   float* __restrict__ out) {
    __shared__ __align__(32) unsigned short lds_x[16 * DIM];   // 32 KB A band
    __shared__ float red[16][2];                               // per-row sum / sumsq

    const int M0    = blockIdx.x * 16;
    const int bbase = M0 & ~(SEQ - 1);       // batch row base
    const int l0    = M0 & (SEQ - 1);        // position inside batch
    const int lane  = threadIdx.x & 31;
    const int wav   = threadIdx.x >> 5;
    const int l16   = lane & 15;
    const int kh    = lane >> 4;
    const int N0    = wav * 128;

    if (threadIdx.x < 32) reinterpret_cast<float*>(red)[threadIdx.x] = 0.0f;

    const uint4* gsrc = reinterpret_cast<const uint4*>(xbf + (size_t)M0 * DIM);
    for (int it = threadIdx.x; it < 2048; it += 256)
        reinterpret_cast<uint4*>(lds_x)[it] = gsrc[it];
    __syncthreads();

    v8f oacc[8];
#pragma unroll
    for (int t = 0; t < 8; ++t) oacc[t] = zero8();

    const int shifts[NGRP] = {0, -32, -16, -4, -1, 1, 4, 16, 32};

#pragma unroll
    for (int j = 0; j < NGRP; ++j) {
        const int s     = shifts[j];
        const int jbase = j * DIM;          // gate_W row offset for this group

        v8f gacc[8];
#pragma unroll
        for (int t = 0; t < 8; ++t) gacc[t] = zero8();

        for (int k = 0; k < DIM; k += 32) {
            v16bf a = *reinterpret_cast<const v16bf*>(&lds_x[l16 * DIM + k + kh * 16]);
            __builtin_prefetch(&gw[(size_t)(jbase + N0 + l16) * DIM + k + 128], 0, 1);
#pragma unroll
            for (int t = 0; t < 8; ++t) {
                int colB = jbase + N0 + t * 16 + l16;
                v16bf b = *reinterpret_cast<const v16bf*>(&gw[(size_t)colB * DIM + k + kh * 16]);
                gacc[t] = __builtin_amdgcn_wmma_f32_16x16x32_bf16(
                              false, a, false, b, (short)0, gacc[t], false, false);
            }
        }

        // sigmoid gate, gather shifted v, accumulate
#pragma unroll
        for (int t = 0; t < 8; ++t) {
            int col = N0 + t * 16 + l16;
#pragma unroll
            for (int i = 0; i < 8; ++i) {
                int lrow = l0 + i + kh * 8;
                int srow = bbase + ((lrow + s + SEQ) & (SEQ - 1));
                float g  = 1.0f / (1.0f + __expf(-gacc[t][i]));
                float vv = bf2f(vbf[(size_t)srow * DIM + col]);
                oacc[t][i] += g * vv;
            }
        }
    }

    // ---- LayerNorm: per-row reduction across the workgroup via LDS atomics --
#pragma unroll
    for (int i = 0; i < 8; ++i) {
        int rowl = i + kh * 8;
        float s1 = 0.0f, s2 = 0.0f;
#pragma unroll
        for (int t = 0; t < 8; ++t) { float o = oacc[t][i]; s1 += o; s2 += o * o; }
        atomicAdd(&red[rowl][0], s1);
        atomicAdd(&red[rowl][1], s2);
    }
    __syncthreads();

#pragma unroll
    for (int t = 0; t < 8; ++t) {
        int col = N0 + t * 16 + l16;
        float ga = gamma[col];
        float be = beta[col];
#pragma unroll
        for (int i = 0; i < 8; ++i) {
            int rowl = i + kh * 8;
            float mu  = red[rowl][0] * (1.0f / DIM);
            float var = red[rowl][1] * (1.0f / DIM) - mu * mu;
            float rs  = rsqrtf(var + 1e-5f);
            out[(size_t)(M0 + rowl) * DIM + col] = (oacc[t][i] - mu) * rs * ga + be;
        }
    }
}

// -------------------------------------------------------------- launcher ----
extern "C" void kernel_launch(void* const* d_in, const int* in_sizes, int n_in,
                              void* d_out, int out_size, void* d_ws, size_t ws_size,
                              hipStream_t stream) {
    const float* x     = (const float*)d_in[0];   // (4,4096,1024)
    const float* gateW = (const float*)d_in[1];   // (9216,1024)
    const float* valW  = (const float*)d_in[2];   // (1024,1024)
    const float* gamma = (const float*)d_in[3];   // (1024,)
    const float* beta  = (const float*)d_in[4];   // (1024,)
    float* out = (float*)d_out;

    const size_t N_X  = (size_t)ROWS * DIM;            // 16,777,216
    const size_t N_GW = (size_t)NGRP * DIM * DIM;      //  9,437,184
    const size_t N_VW = (size_t)DIM * DIM;             //  1,048,576

    char* ws = (char*)d_ws;                            // needs ~84 MB
    unsigned short* xbf  = (unsigned short*)(ws);
    unsigned short* gwbf = (unsigned short*)(ws + N_X * 2);
    unsigned short* wvbf = (unsigned short*)(ws + N_X * 2 + N_GW * 2);
    unsigned short* vbf  = (unsigned short*)(ws + N_X * 2 + N_GW * 2 + N_VW * 2);

    cvt_f32_bf16<<<4096, 256, 0, stream>>>(x,     xbf,  (int)(N_X  / 4));
    cvt_f32_bf16<<<4096, 256, 0, stream>>>(gateW, gwbf, (int)(N_GW / 4));
    cvt_f32_bf16<<<1024, 256, 0, stream>>>(valW,  wvbf, (int)(N_VW / 4));

    value_gemm<<<ROWS / 16, 256, 0, stream>>>(xbf, wvbf, vbf);
    gate_mix_ln<<<ROWS / 16, 256, 0, stream>>>(xbf, gwbf, vbf, gamma, beta, out);
}
